// NunchakuWanAttention_16097537425973
// MI455X (gfx1250) — matrix operand order repaired and checked by reading, not compiled, not run on hardware
//
#include <hip/hip_runtime.h>

// ---------------------------------------------------------------------------
// Types for CDNA5 WMMA (gfx1250, wave32)
// ---------------------------------------------------------------------------
typedef __attribute__((ext_vector_type(16))) __bf16 v16bf;
typedef __attribute__((ext_vector_type(8)))  float  v8f;

union ABFrag {
  v16bf v;
  uint4 u4[2];
  unsigned short h[16];
};

__device__ __forceinline__ unsigned short f32_to_bf16(float f) {
  unsigned int u = __float_as_uint(f);
  u += 0x7FFFu + ((u >> 16) & 1u);   // round to nearest even
  return (unsigned short)(u >> 16);
}

__device__ __forceinline__ v8f wmma_bf16(const ABFrag& a, const ABFrag& b, v8f c) {
  // D = A(16x32) * B(32x16) + C(16x16 f32)
  return __builtin_amdgcn_wmma_f32_16x16x32_bf16(
      /*neg_a=*/false, a.v, /*neg_b=*/false, b.v,
      /*c_mod=*/(short)0, c, /*reuse_a=*/false, /*reuse_b=*/false);
}

#define S_LEN 4096
#define DIM   1536
#define NHEAD 12
#define DH    128
#define FQKV  4608

// ---------------------------------------------------------------------------
// Kernel 0: fp32 -> bf16 conversion
// ---------------------------------------------------------------------------
__global__ void cvt_f32_bf16(const float* __restrict__ in,
                             unsigned short* __restrict__ out, int n) {
  int i = blockIdx.x * blockDim.x + threadIdx.x;
  if (i < n) out[i] = f32_to_bf16(in[i]);
}

// ---------------------------------------------------------------------------
// Kernel 1/4: C[M,N] = A[M,K] * B[N,K]^T + bias[N]   (bf16 in, f32 out)
// 128-thread block computes a 128x128 tile; A/B k-slices (128x32) staged into
// double-buffered LDS with GLOBAL_LOAD_ASYNC_TO_LDS_B128, pipelined vs WMMA.
// ---------------------------------------------------------------------------
__global__ __launch_bounds__(128) void gemm_bf16_nt_async(
    const unsigned short* __restrict__ A, const unsigned short* __restrict__ B,
    const float* __restrict__ bias, float* __restrict__ C,
    int M, int N, int K) {
  const int t    = threadIdx.x;   // 0..127
  const int lane = t & 31;
  const int l16  = lane & 15;
  const int half = lane >> 4;
  const int w    = t >> 5;        // wave in block
  const int wm   = w >> 1;        // quadrant row (0..1)
  const int wn   = w & 1;         // quadrant col (0..1)

  const int tilesN = N >> 7;
  const int tmB = blockIdx.x / tilesN;
  const int tnB = blockIdx.x % tilesN;

  __shared__ unsigned short As[2][128 * 32];   // 8KB per buffer
  __shared__ unsigned short Bs[2][128 * 32];   // 8KB per buffer

  const unsigned short* gArow = A + (size_t)(tmB * 128 + t) * K;
  const unsigned short* gBrow = B + (size_t)(tnB * 128 + t) * K;

  auto stage = [&](int buf, int kb) {
    const unsigned short* ga = gArow + kb;
    const unsigned short* gb = gBrow + kb;
    unsigned int la = (unsigned int)(size_t)&As[buf][t * 32];
    unsigned int lb = (unsigned int)(size_t)&Bs[buf][t * 32];
#pragma unroll
    for (int c = 0; c < 4; ++c) {
      asm volatile("global_load_async_to_lds_b128 %0, %1, off"
                   :: "v"(la + c * 16), "v"(ga + c * 8) : "memory");
      asm volatile("global_load_async_to_lds_b128 %0, %1, off"
                   :: "v"(lb + c * 16), "v"(gb + c * 8) : "memory");
    }
  };

  const v8f vz = {0.f, 0.f, 0.f, 0.f, 0.f, 0.f, 0.f, 0.f};
  v8f acc[4][4];
#pragma unroll
  for (int i = 0; i < 4; ++i)
#pragma unroll
    for (int j = 0; j < 4; ++j) acc[i][j] = vz;

  const int NKs = K >> 5;
  stage(0, 0);

  for (int kt = 0; kt < NKs; ++kt) {
    if (kt + 1 < NKs) {
      stage((kt + 1) & 1, (kt + 1) * 32);
      asm volatile("s_wait_asynccnt 0x8" ::: "memory");
    } else {
      asm volatile("s_wait_asynccnt 0x0" ::: "memory");
    }
    __syncthreads();

    const unsigned short* Ab = As[kt & 1];
    const unsigned short* Bb = Bs[kt & 1];
    ABFrag a[4], b[4];
#pragma unroll
    for (int i = 0; i < 4; ++i) {
      const unsigned short* p = Ab + (wm * 64 + i * 16 + l16) * 32;
      a[i].u4[0] = *(const uint4*)(p + half * 8);
      a[i].u4[1] = *(const uint4*)(p + 16 + half * 8);
    }
#pragma unroll
    for (int j = 0; j < 4; ++j) {
      const unsigned short* p = Bb + (wn * 64 + j * 16 + l16) * 32;
      b[j].u4[0] = *(const uint4*)(p + half * 8);
      b[j].u4[1] = *(const uint4*)(p + 16 + half * 8);
    }
#pragma unroll
    for (int i = 0; i < 4; ++i)
#pragma unroll
      for (int j = 0; j < 4; ++j)
        acc[i][j] = wmma_bf16(a[i], b[j], acc[i][j]);

    __syncthreads();
  }

#pragma unroll
  for (int i = 0; i < 4; ++i)
#pragma unroll
    for (int j = 0; j < 4; ++j) {
      const int col = tnB * 128 + wn * 64 + j * 16 + l16;
      const float bv = bias[col];
#pragma unroll
      for (int g = 0; g < 8; ++g) {
        const int row = tmB * 128 + wm * 64 + i * 16 + g + half * 8;
        C[(size_t)row * N + col] = acc[i][j][g] + bv;
      }
    }
}

// ---------------------------------------------------------------------------
// Kernel 2: per-row RMSNorm(q,k) + RoPE(q,k) + bf16 pack of q,k,v
// ---------------------------------------------------------------------------
__global__ __launch_bounds__(256) void norm_rope_pack(
    const float* __restrict__ qkv,
    const float* __restrict__ nqw, const float* __restrict__ nkw,
    const float* __restrict__ fc, const float* __restrict__ fs,
    unsigned short* __restrict__ qb, unsigned short* __restrict__ kb,
    unsigned short* __restrict__ vb) {
  const int s = blockIdx.x;
  const int t = threadIdx.x;
  const float* row = qkv + (size_t)s * FQKV;
  __shared__ float red[16];

  float sq = 0.f, sk = 0.f;
  for (int i = t; i < DIM; i += 256) {
    float x = row[i];        sq += x * x;
    float y = row[DIM + i];  sk += y * y;
  }
#pragma unroll
  for (int m = 16; m >= 1; m >>= 1) {
    sq += __shfl_xor(sq, m);
    sk += __shfl_xor(sk, m);
  }
  const int w = t >> 5;
  if ((t & 31) == 0) { red[w] = sq; red[8 + w] = sk; }
  __syncthreads();
  float sumq = 0.f, sumk = 0.f;
#pragma unroll
  for (int i = 0; i < 8; ++i) { sumq += red[i]; sumk += red[8 + i]; }
  const float rq = rsqrtf(sumq * (1.0f / DIM) + 1e-6f);
  const float rk = rsqrtf(sumk * (1.0f / DIM) + 1e-6f);

  for (int p = t; p < DIM / 2; p += 256) {
    const int d  = 2 * p;
    const int dh = d & (DH - 1);
    const float c  = fc[(size_t)s * DH + dh];       // freqs_cos[..., 0::2]
    const float sn = fs[(size_t)s * DH + dh + 1];   // freqs_sin[..., 1::2]

    float x1 = row[d]     * rq * nqw[d];
    float x2 = row[d + 1] * rq * nqw[d + 1];
    qb[(size_t)s * DIM + d]     = f32_to_bf16(x1 * c - x2 * sn);
    qb[(size_t)s * DIM + d + 1] = f32_to_bf16(x1 * sn + x2 * c);

    float y1 = row[DIM + d]     * rk * nkw[d];
    float y2 = row[DIM + d + 1] * rk * nkw[d + 1];
    kb[(size_t)s * DIM + d]     = f32_to_bf16(y1 * c - y2 * sn);
    kb[(size_t)s * DIM + d + 1] = f32_to_bf16(y1 * sn + y2 * c);
  }
  for (int i = t; i < DIM; i += 256)
    vb[(size_t)s * DIM + i] = f32_to_bf16(row[2 * DIM + i]);
}

// ---------------------------------------------------------------------------
// Kernel 3: flash attention. Block = 4 waves = 64 q-rows of one head.
// K tile (32x128) async-copied to LDS; V tile loaded + stored transposed in
// LDS so every WMMA B-fragment is two contiguous ds_load_b128. Double-
// buffered, pipelined with s_wait_asynccnt.
// ---------------------------------------------------------------------------
__global__ __launch_bounds__(128) void flash_attn(
    const unsigned short* __restrict__ Qb, const unsigned short* __restrict__ Kb,
    const unsigned short* __restrict__ Vb, unsigned short* __restrict__ Ob) {
  const int t    = threadIdx.x;
  const int lane = t & 31;
  const int l16  = lane & 15;
  const int half = lane >> 4;
  const int w    = t >> 5;               // wave in block: q tile
  const int head = blockIdx.x >> 6;      // 64 q-blocks per head
  const int qblk = blockIdx.x & 63;
  const int hoff = head * DH;
  const int qrow0 = qblk * 64 + w * 16;

  __shared__ unsigned short Ks[2][32 * 128];   // 8KB/buf, K tile row-major
  __shared__ unsigned short Vt[2][128 * 32];   // 8KB/buf, V tile transposed
  __shared__ unsigned short Pl[4][16 * 32];    // per-wave P staging
  unsigned short* P = Pl[w];

  const int srow = t >> 2;   // 0..31: K/V source row handled by this thread
  const int c4   = t & 3;    // 32-wide column chunk

  auto stage = [&](int buf, int kb0) {
    // K tile: per-lane async b128 copies into LDS (row-major 32x128)
    {
      const unsigned short* g = Kb + (size_t)(kb0 + srow) * DIM + hoff + c4 * 32;
      unsigned int l = (unsigned int)(size_t)&Ks[buf][srow * 128 + c4 * 32];
#pragma unroll
      for (int c = 0; c < 4; ++c)
        asm volatile("global_load_async_to_lds_b128 %0, %1, off"
                     :: "v"(l + c * 16), "v"(g + c * 8) : "memory");
    }
    // V tile: contiguous global loads, transposed stores Vt[d][r]
    {
      const unsigned short* g = Vb + (size_t)(kb0 + srow) * DIM + hoff + c4 * 32;
      ABFrag v0, v1;
      v0.u4[0] = *(const uint4*)(g);
      v0.u4[1] = *(const uint4*)(g + 8);
      v1.u4[0] = *(const uint4*)(g + 16);
      v1.u4[1] = *(const uint4*)(g + 24);
      unsigned short* dst = &Vt[buf][0];
#pragma unroll
      for (int e = 0; e < 16; ++e) {
        dst[(c4 * 32 + e) * 32 + srow]      = v0.h[e];
        dst[(c4 * 32 + 16 + e) * 32 + srow] = v1.h[e];
      }
    }
  };

  // Q fragments: 16 rows x 128 (4 k-chunks of 32)
  ABFrag qa[4];
  {
    const unsigned short* qrow = Qb + (size_t)(qrow0 + l16) * DIM + hoff;
#pragma unroll
    for (int dc = 0; dc < 4; ++dc) {
      qa[dc].u4[0] = *(const uint4*)(qrow + dc * 32 + half * 8);
      qa[dc].u4[1] = *(const uint4*)(qrow + dc * 32 + 16 + half * 8);
    }
  }

  const v8f vz = {0.f, 0.f, 0.f, 0.f, 0.f, 0.f, 0.f, 0.f};
  v8f o[8];
#pragma unroll
  for (int dt = 0; dt < 8; ++dt) o[dt] = vz;
  float rm[8], rs[8];
#pragma unroll
  for (int g = 0; g < 8; ++g) { rm[g] = -__builtin_inff(); rs[g] = 0.f; }

  const float sc2 = 0.08838834764831845f * 1.4426950408889634f;  // scale*log2e
  const int NKT = S_LEN / 32;   // 128

  stage(0, 0);

  for (int kt = 0; kt < NKT; ++kt) {
    const int buf = kt & 1;
    if (kt + 1 < NKT) {
      stage(buf ^ 1, (kt + 1) * 32);
      asm volatile("s_wait_asynccnt 0x4" ::: "memory");
    } else {
      asm volatile("s_wait_asynccnt 0x0" ::: "memory");
    }
    __syncthreads();   // K async + V transpose of this buffer are visible

    // ---- scores S = Q * K^T (16x32 tile) ----
    v8f s0 = vz, s1 = vz;
#pragma unroll
    for (int nt = 0; nt < 2; ++nt) {
      const unsigned short* krow = &Ks[buf][(nt * 16 + l16) * 128 + half * 8];
#pragma unroll
      for (int dc = 0; dc < 4; ++dc) {
        ABFrag b;
        b.u4[0] = *(const uint4*)(krow + dc * 32);
        b.u4[1] = *(const uint4*)(krow + dc * 32 + 16);
        if (nt == 0) s0 = wmma_bf16(qa[dc], b, s0);
        else         s1 = wmma_bf16(qa[dc], b, s1);
      }
    }

    // ---- online softmax (row g + 8*half lives in VGPR g) ----
    float p0[8], p1[8];
#pragma unroll
    for (int g = 0; g < 8; ++g) {
      float mx = fmaxf(s0[g], s1[g]);
#pragma unroll
      for (int m = 8; m >= 1; m >>= 1) mx = fmaxf(mx, __shfl_xor(mx, m));
      const float nm = fmaxf(rm[g], mx * sc2);
      const float f  = exp2f(rm[g] - nm);
      rm[g] = nm;
      rs[g] *= f;
#pragma unroll
      for (int dt = 0; dt < 8; ++dt) o[dt][g] *= f;
      const float a = exp2f(s0[g] * sc2 - nm);
      const float b = exp2f(s1[g] * sc2 - nm);
      p0[g] = a; p1[g] = b;
      float sum = a + b;
#pragma unroll
      for (int m = 8; m >= 1; m >>= 1) sum += __shfl_xor(sum, m);
      rs[g] += sum;
    }

    // ---- P: C-layout -> A-layout via per-wave LDS staging ----
#pragma unroll
    for (int g = 0; g < 8; ++g) {
      const int rowp = (g + half * 8) * 32;
      P[rowp + l16]      = f32_to_bf16(p0[g]);
      P[rowp + 16 + l16] = f32_to_bf16(p1[g]);
    }
    asm volatile("s_wait_dscnt 0" ::: "memory");
    ABFrag pa;
    {
      const unsigned short* prow = P + l16 * 32;
      pa.u4[0] = *(const uint4*)(prow + half * 8);
      pa.u4[1] = *(const uint4*)(prow + 16 + half * 8);
    }

    // ---- O += P * V : B frags contiguous from transposed V ----
#pragma unroll
    for (int dt = 0; dt < 8; ++dt) {
      ABFrag bv;
      const unsigned short* vrow = &Vt[buf][(dt * 16 + l16) * 32 + half * 8];
      bv.u4[0] = *(const uint4*)(vrow);
      bv.u4[1] = *(const uint4*)(vrow + 16);
      o[dt] = wmma_bf16(pa, bv, o[dt]);
    }

    __syncthreads();   // done reading this buffer before it is re-staged
  }

#pragma unroll
  for (int g = 0; g < 8; ++g) {
    const float inv = 1.0f / rs[g];
    const int row = qrow0 + g + half * 8;
#pragma unroll
    for (int dt = 0; dt < 8; ++dt)
      Ob[(size_t)row * DIM + hoff + dt * 16 + l16] = f32_to_bf16(o[dt][g] * inv);
  }
}

// ---------------------------------------------------------------------------
// Host-side orchestration
// ---------------------------------------------------------------------------
extern "C" void kernel_launch(void* const* d_in, const int* in_sizes, int n_in,
                              void* d_out, int out_size, void* d_ws, size_t ws_size,
                              hipStream_t stream) {
  const float* hidden = (const float*)d_in[0];
  const float* w_qkv  = (const float*)d_in[1];
  const float* b_qkv  = (const float*)d_in[2];
  const float* nqw    = (const float*)d_in[3];
  const float* nkw    = (const float*)d_in[4];
  const float* w_out  = (const float*)d_in[5];
  const float* b_out  = (const float*)d_in[6];
  const float* fc     = (const float*)d_in[7];
  const float* fs     = (const float*)d_in[8];
  float* out = (float*)d_out;

  char* ws = (char*)d_ws;
  size_t off = 0;
  auto alloc = [&](size_t bytes) -> void* {
    void* p = ws + off;
    off += (bytes + 255) & ~(size_t)255;
    return p;
  };

  float*          qkv = (float*)         alloc((size_t)S_LEN * FQKV * 4);
  unsigned short* hb  = (unsigned short*)alloc((size_t)S_LEN * DIM * 2);
  unsigned short* wqb = (unsigned short*)alloc((size_t)FQKV * DIM * 2);
  unsigned short* wob = (unsigned short*)alloc((size_t)DIM * DIM * 2);
  unsigned short* qb  = (unsigned short*)alloc((size_t)S_LEN * DIM * 2);
  unsigned short* kbf = (unsigned short*)alloc((size_t)S_LEN * DIM * 2);
  unsigned short* vbf = (unsigned short*)alloc((size_t)S_LEN * DIM * 2);
  unsigned short* ab  = (unsigned short*)alloc((size_t)S_LEN * DIM * 2);

  // 0) fp32 -> bf16 conversions
  {
    int n = S_LEN * DIM;
    cvt_f32_bf16<<<(n + 255) / 256, 256, 0, stream>>>(hidden, hb, n);
  }
  {
    int n = FQKV * DIM;
    cvt_f32_bf16<<<(n + 255) / 256, 256, 0, stream>>>(w_qkv, wqb, n);
  }
  {
    int n = DIM * DIM;
    cvt_f32_bf16<<<(n + 255) / 256, 256, 0, stream>>>(w_out, wob, n);
  }

  // 1) qkv = hidden @ w_qkv^T + b_qkv   (M=4096, N=4608, K=1536)
  {
    int blocks = (S_LEN / 128) * (FQKV / 128);  // 32*36 = 1152
    gemm_bf16_nt_async<<<blocks, 128, 0, stream>>>(hb, wqb, b_qkv, qkv,
                                                   S_LEN, FQKV, DIM);
  }

  // 2) RMSNorm(q,k) + RoPE + pack q/k/v to bf16
  norm_rope_pack<<<S_LEN, 256, 0, stream>>>(qkv, nqw, nkw, fc, fs, qb, kbf, vbf);

  // 3) flash attention: 12 heads * 64 q-blocks = 768 blocks (4 waves each)
  flash_attn<<<NHEAD * (S_LEN / 64), 128, 0, stream>>>(qb, kbf, vbf, ab);

  // 4) out = attn @ w_out^T + b_out   (M=4096, N=1536, K=1536)
  {
    int blocks = (S_LEN / 128) * (DIM / 128);   // 32*12 = 384
    gemm_bf16_nt_async<<<blocks, 128, 0, stream>>>(ab, wob, b_out, out,
                                                   S_LEN, DIM, DIM);
  }
}
